// TransformerBlock_9002251453084
// MI455X (gfx1250) — matrix operand order repaired and checked by reading, not compiled, not run on hardware
//
#include <hip/hip_runtime.h>

// ---------------------------------------------------------------------------
// Transformer block for MI455X (gfx1250): bf16 WMMA GEMMs + flash attention.
// Matrix math: V_WMMA_F32_16X16X32_BF16 (wave32, f32 accumulation).
// Data movement: GLOBAL_LOAD_ASYNC_TO_LDS_B128 + S_WAIT_ASYNCCNT (CDNA5 async
// path), double-buffered LDS staging in the GEMM.
// ---------------------------------------------------------------------------

#define DMODEL 1024
#define NHEADS 16
#define HDIM   64
#define DFF    4096
#define SEQ    2048
#define BATCH  4
#define MROWS  (BATCH * SEQ)

typedef __attribute__((ext_vector_type(16))) __bf16 v16bf;
typedef __attribute__((ext_vector_type(8)))  float  v8f;

__device__ __forceinline__ v8f wmma_bf16f32(v16bf a, v16bf b, v8f c) {
  // 8 args: (neg_a, A, neg_b, B, c_mod, C, reuse_a, reuse_b)
  return __builtin_amdgcn_wmma_f32_16x16x32_bf16(false, a, false, b, (short)0, c,
                                                 false, false);
}

// Async copy of 16 bytes global -> LDS (VGLOBAL encoding, GV mode, ASYNCcnt).
__device__ __forceinline__ void async_b128(unsigned lds_byte_off, const __bf16* g) {
  asm volatile("global_load_async_to_lds_b128 %0, %1, off"
               :: "v"(lds_byte_off), "v"(g)
               : "memory");
}
__device__ __forceinline__ void wait_async0() {
  asm volatile("s_wait_asynccnt 0x0" ::: "memory");
}

// A fragment (16x32 MxK, 16-bit): lane row r, element j -> K = 16*hf + j.
// Source: p[r*ld + k], row-major.
__device__ __forceinline__ v16bf load_frag_a(const __bf16* p, int ld, int lane) {
  const int hf = (lane >> 4) & 1, r = lane & 15;
  v16bf a;
#pragma unroll
  for (int j = 0; j < 16; ++j) {
    const int k = 16 * hf + j;
    a[j] = p[r * ld + k];
  }
  return a;
}

// B fragment (32x16 KxN, 16-bit): lane n = lane&15, element j -> K = 16*hf + j.
// Element (k,n) located at p[n*ld + k]  (N-major storage).
__device__ __forceinline__ v16bf load_frag_b_nmajor(const __bf16* p, int ld, int lane) {
  const int hf = (lane >> 4) & 1, n = lane & 15;
  v16bf b;
#pragma unroll
  for (int j = 0; j < 16; ++j) b[j] = p[n * ld + 16 * hf + j];
  return b;
}

// Element (k,n) located at p[k*ld + n]  (K-major storage).
__device__ __forceinline__ v16bf load_frag_b_kmajor(const __bf16* p, int ld, int lane) {
  const int hf = (lane >> 4) & 1, n = lane & 15;
  v16bf b;
#pragma unroll
  for (int j = 0; j < 16; ++j) b[j] = p[(16 * hf + j) * ld + n];
  return b;
}

// ---------------------------------------------------------------------------
// Tiled GEMM: C[M,N](f32) = A[M,K](bf16,row-major) * W[N,K](bf16,row-major)^T
// EPI==1 fuses "+ Res[M,N]".  Block tile 128x128, BK=32, 8 waves of 32x64.
// Double-buffered LDS, filled with global_load_async_to_lds_b128.
// Dynamic LDS only => region starts at LDS byte offset 0.
// ---------------------------------------------------------------------------
#define BM 128
#define BN 128
#define BK 32
#define LDS_STRIDE 40                       // 32 + pad (8B-aligned rows)
#define ATILE_ELEMS (BM * LDS_STRIDE)       // 5120 bf16
#define ATILE_BYTES (ATILE_ELEMS * 2)       // 10240
#define GBUF_ELEMS  (2 * ATILE_ELEMS)       // A-tile + B-tile
#define GBUF_BYTES  (GBUF_ELEMS * 2)        // 20480
#define GEMM_LDS_BYTES (2 * GBUF_BYTES)     // ping-pong: 40960

__device__ __forceinline__ void gemm_stage_async(const __bf16* __restrict__ A,
                                                 const __bf16* __restrict__ W,
                                                 int K, int bm, int bn, int k0,
                                                 unsigned bufbase, int tid) {
  // 128x32 bf16 per tile = 512 x 16B chunks per tile; 2 tiles / 256 threads.
#pragma unroll
  for (int i = tid; i < 512; i += 256) {
    const int row = i >> 2, col = (i & 3) * 8;
    const unsigned loff = bufbase + (unsigned)(row * LDS_STRIDE + col) * 2u;
    async_b128(loff, &A[(size_t)(bm + row) * K + k0 + col]);
    async_b128(loff + ATILE_BYTES, &W[(size_t)(bn + row) * K + k0 + col]);
  }
}

template <int EPI>
__global__ __launch_bounds__(256) void gemm_bf16(
    const __bf16* __restrict__ A, const __bf16* __restrict__ W,
    float* __restrict__ C, const float* __restrict__ Res, int M, int N, int K) {
  extern __shared__ __bf16 smem[];  // 2 x (A-tile | B-tile)

  const int tid = threadIdx.x;
  const int lane = tid & 31, wid = tid >> 5;
  const int bm = blockIdx.y * BM, bn = blockIdx.x * BN;
  const int wm = (wid & 3) * 32;   // 4 waves along M, 32 rows each
  const int wn = (wid >> 2) * 64;  // 2 waves along N, 64 cols each

  v8f acc[2][4];
#pragma unroll
  for (int i = 0; i < 2; ++i)
#pragma unroll
    for (int j = 0; j < 4; ++j)
#pragma unroll
      for (int e = 0; e < 8; ++e) acc[i][j][e] = 0.0f;

  // Prologue: fill buffer 0.
  gemm_stage_async(A, W, K, bm, bn, 0, 0u, tid);
  wait_async0();
  __syncthreads();

  unsigned buf = 0;
  for (int k0 = 0; k0 < K; k0 += BK) {
    // Kick off next tile into the other buffer while we compute this one.
    if (k0 + BK < K)
      gemm_stage_async(A, W, K, bm, bn, k0 + BK, (buf ^ 1u) * GBUF_BYTES, tid);

    const __bf16* Asb = smem + buf * GBUF_ELEMS;
    const __bf16* Bsb = Asb + ATILE_ELEMS;

    v16bf af[2], bfg[4];
#pragma unroll
    for (int tm = 0; tm < 2; ++tm)
      af[tm] = load_frag_a(&Asb[(wm + tm * 16) * LDS_STRIDE], LDS_STRIDE, lane);
#pragma unroll
    for (int tn = 0; tn < 4; ++tn)
      bfg[tn] = load_frag_b_nmajor(&Bsb[(wn + tn * 16) * LDS_STRIDE], LDS_STRIDE, lane);
#pragma unroll
    for (int tm = 0; tm < 2; ++tm)
#pragma unroll
      for (int tn = 0; tn < 4; ++tn)
        acc[tm][tn] = wmma_bf16f32(af[tm], bfg[tn], acc[tm][tn]);

    wait_async0();     // this wave's async fills of buf^1 done
    __syncthreads();   // everyone done reading buf + filling buf^1
    buf ^= 1u;
  }

  // C layout: VGPR r -> row r + 8*hf, lane&15 -> column
  const int hf = (lane >> 4) & 1, cn = lane & 15;
#pragma unroll
  for (int tm = 0; tm < 2; ++tm)
#pragma unroll
    for (int tn = 0; tn < 4; ++tn)
#pragma unroll
      for (int r = 0; r < 8; ++r) {
        const int row = bm + wm + tm * 16 + r + 8 * hf;
        const int col = bn + wn + tn * 16 + cn;
        float v = acc[tm][tn][r];
        if (EPI == 1) v += Res[(size_t)row * N + col];
        C[(size_t)row * N + col] = v;
      }
}

// ---------------------------------------------------------------------------
// RMSNorm: f32 in -> bf16 out (one workgroup per row of 1024)
// ---------------------------------------------------------------------------
__global__ __launch_bounds__(256) void rmsnorm_bf16(const float* __restrict__ x,
                                                    const float* __restrict__ g,
                                                    __bf16* __restrict__ out) {
  const int row = blockIdx.x, tid = threadIdx.x;
  const int lane = tid & 31, wid = tid >> 5;
  float s = 0.f;
  for (int i = tid; i < DMODEL; i += 256) {
    const float v = x[(size_t)row * DMODEL + i];
    s += v * v;
  }
#pragma unroll
  for (int off = 16; off > 0; off >>= 1) s += __shfl_xor(s, off, 32);
  __shared__ float red[9];
  if (lane == 0) red[wid] = s;
  __syncthreads();
  if (tid == 0) {
    float t = 0.f;
    for (int j = 0; j < 8; ++j) t += red[j];
    red[8] = rsqrtf(t / (float)DMODEL + 1e-5f);
  }
  __syncthreads();
  const float rinv = red[8];
  for (int i = tid; i < DMODEL; i += 256)
    out[(size_t)row * DMODEL + i] = (__bf16)(x[(size_t)row * DMODEL + i] * g[i] * rinv);
}

// ---------------------------------------------------------------------------
// Fused RoPE + scatter qkv(f32, [B*S, 3072]) -> qT/kT/vT (bf16, [B,H,S,64])
// ---------------------------------------------------------------------------
__global__ __launch_bounds__(256) void rope_scatter(const float* __restrict__ qkv,
                                                    __bf16* __restrict__ qT,
                                                    __bf16* __restrict__ kT,
                                                    __bf16* __restrict__ vT) {
  const int id = blockIdx.x * 256 + threadIdx.x;  // B*S*H*32 threads
  const int b = id >> 20;
  const int s = (id >> 9) & (SEQ - 1);
  const int h = (id >> 5) & (NHEADS - 1);
  const int t = id & 31;
  const size_t rowq = (size_t)(b * SEQ + s) * (3 * DMODEL);
  const int c = h * HDIM + 2 * t;
  const float q1 = qkv[rowq + c], q2 = qkv[rowq + c + 1];
  const float k1 = qkv[rowq + DMODEL + c], k2 = qkv[rowq + DMODEL + c + 1];
  const float v1 = qkv[rowq + 2 * DMODEL + c], v2 = qkv[rowq + 2 * DMODEL + c + 1];
  const float inv_freq = __expf(-0.28782313662425572f * (float)t);  // 10000^(-t/32)
  float sn, cs;
  __sincosf((float)s * inv_freq, &sn, &cs);
  const size_t o = ((size_t)((b * NHEADS + h) * SEQ + s)) * HDIM + 2 * t;
  qT[o]     = (__bf16)(q1 * cs - q2 * sn);
  qT[o + 1] = (__bf16)(q2 * cs + q1 * sn);
  kT[o]     = (__bf16)(k1 * cs - k2 * sn);
  kT[o + 1] = (__bf16)(k2 * cs + k1 * sn);
  vT[o]     = (__bf16)v1;
  vT[o + 1] = (__bf16)v2;
}

// ---------------------------------------------------------------------------
// Flash attention (causal): 8 waves x 16 q-rows per block, 32-key KV tiles
// staged with async-to-LDS; scores and P*V on bf16 WMMA; online softmax.
// Dynamic LDS layout (byte offsets): Kt @ 0, Vt @ 4608, Pw @ 9216 (1152/wave).
// ---------------------------------------------------------------------------
#define KT_LD 72
#define PW_LD 36
#define KT_BYTES (32 * KT_LD * 2)                    // 4608
#define PW_BASE  (2 * KT_BYTES)                      // 9216
#define ATTN_LDS_BYTES (PW_BASE + 8 * 16 * PW_LD * 2)  // 18432

__global__ __launch_bounds__(256) void flash_attn(const __bf16* __restrict__ qT,
                                                  const __bf16* __restrict__ kT,
                                                  const __bf16* __restrict__ vT,
                                                  __bf16* __restrict__ attn) {
  extern __shared__ __bf16 smem[];
  __bf16* Kt = smem;                       // [key][dim], ld=KT_LD
  __bf16* Vt = smem + 32 * KT_LD;          // [key][dim], ld=KT_LD
  const int tid = threadIdx.x, lane = tid & 31, wid = tid >> 5;
  __bf16* Pw = smem + 2 * 32 * KT_LD + wid * 16 * PW_LD;  // per-wave P scratch

  const int bh = blockIdx.x / (SEQ / 128);
  const int qbase = (blockIdx.x % (SEQ / 128)) * 128;
  const int b = bh >> 4, h = bh & 15;

  const __bf16* Q  = qT + (size_t)bh * SEQ * HDIM;
  const __bf16* Kp = kT + (size_t)bh * SEQ * HDIM;
  const __bf16* Vp = vT + (size_t)bh * SEQ * HDIM;

  const int q0 = qbase + wid * 16;

  // Q fragments (row-major, ld=64): dims 0..31 and 32..63
  const v16bf qa0 = load_frag_a(Q + (size_t)q0 * HDIM, HDIM, lane);
  const v16bf qa1 = load_frag_a(Q + (size_t)q0 * HDIM + 32, HDIM, lane);

  v8f o[4];
  float m_i[8], l_i[8];
#pragma unroll
  for (int nc = 0; nc < 4; ++nc)
#pragma unroll
    for (int e = 0; e < 8; ++e) o[nc][e] = 0.f;
#pragma unroll
  for (int r = 0; r < 8; ++r) { m_i[r] = -1e30f; l_i[r] = 0.f; }

  const int hf = (lane >> 4) & 1, cn = lane & 15;
  const int ktmax = (qbase + 127) >> 5;

  for (int kt = 0; kt <= ktmax; ++kt) {
    const int kstart = kt << 5;
    __syncthreads();  // previous iteration's reads of Kt/Vt complete
    {  // async-stage 32x64 K and V tiles (16B per thread each)
      const int row = tid >> 3, col = (tid & 7) * 8;
      const unsigned loff = (unsigned)(row * KT_LD + col) * 2u;
      async_b128(loff, &Kp[(size_t)(kstart + row) * HDIM + col]);
      async_b128(loff + KT_BYTES, &Vp[(size_t)(kstart + row) * HDIM + col]);
      wait_async0();
    }
    __syncthreads();

    if (kstart <= q0 + 15) {  // wave-uniform: EXEC all-ones or all-zeros
      v8f s0, s1;
#pragma unroll
      for (int e = 0; e < 8; ++e) { s0[e] = 0.f; s1[e] = 0.f; }
      // B = K^T: element (dim, key) at Kt[key*KT_LD + dim] -> n-major
      const v16bf kb00 = load_frag_b_nmajor(&Kt[0], KT_LD, lane);               // keys 0-15,  dims 0-31
      const v16bf kb01 = load_frag_b_nmajor(&Kt[32], KT_LD, lane);              // keys 0-15,  dims 32-63
      const v16bf kb10 = load_frag_b_nmajor(&Kt[16 * KT_LD], KT_LD, lane);      // keys 16-31, dims 0-31
      const v16bf kb11 = load_frag_b_nmajor(&Kt[16 * KT_LD + 32], KT_LD, lane); // keys 16-31, dims 32-63
      s0 = wmma_bf16f32(qa0, kb00, s0);
      s0 = wmma_bf16f32(qa1, kb01, s0);
      s1 = wmma_bf16f32(qa0, kb10, s1);
      s1 = wmma_bf16f32(qa1, kb11, s1);

#pragma unroll
      for (int r = 0; r < 8; ++r) {
        const int qrow = q0 + r + 8 * hf;
        float e0 = s0[r] * 0.125f;  // 1/sqrt(64)
        float e1 = s1[r] * 0.125f;
        if (kstart + cn > qrow)      e0 = -1e30f;
        if (kstart + 16 + cn > qrow) e1 = -1e30f;
        float mx = fmaxf(e0, e1);
#pragma unroll
        for (int off = 8; off > 0; off >>= 1) mx = fmaxf(mx, __shfl_xor(mx, off, 32));
        const float mn = fmaxf(m_i[r], mx);
        const float corr = (m_i[r] < -1e29f) ? 0.f : __expf(m_i[r] - mn);
        const float p0 = (e0 < -1e29f) ? 0.f : __expf(e0 - mn);
        const float p1 = (e1 < -1e29f) ? 0.f : __expf(e1 - mn);
        float rs = p0 + p1;
#pragma unroll
        for (int off = 8; off > 0; off >>= 1) rs += __shfl_xor(rs, off, 32);
        l_i[r] = l_i[r] * corr + rs;
        m_i[r] = mn;
#pragma unroll
        for (int nc = 0; nc < 4; ++nc) o[nc][r] *= corr;
        // C-fragment -> row-major P (16x32) so it reloads as an A-fragment
        Pw[(r + 8 * hf) * PW_LD + cn]      = (__bf16)p0;
        Pw[(r + 8 * hf) * PW_LD + 16 + cn] = (__bf16)p1;
      }

      // O += P(16x32) * V(32x64); V element (key, dim) at Vt[key*KT_LD+dim]
      const v16bf pa = load_frag_a(Pw, PW_LD, lane);
#pragma unroll
      for (int nc = 0; nc < 4; ++nc) {
        const v16bf vb = load_frag_b_kmajor(&Vt[nc * 16], KT_LD, lane);
        o[nc] = wmma_bf16f32(pa, vb, o[nc]);
      }
    }
  }

#pragma unroll
  for (int r = 0; r < 8; ++r) {
    const float inv = 1.0f / fmaxf(l_i[r], 1e-20f);
    const int qrow = q0 + r + 8 * hf;
#pragma unroll
    for (int nc = 0; nc < 4; ++nc) {
      const float val = o[nc][r] * inv;
      attn[(size_t)(b * SEQ + qrow) * DMODEL + h * HDIM + nc * 16 + cn] = (__bf16)val;
    }
  }
}

// ---------------------------------------------------------------------------
// SwiGLU gate: gated = sigmoid(x1) * x1 * x3  (f32 -> bf16)
// ---------------------------------------------------------------------------
__global__ __launch_bounds__(256) void swiglu_gate(const float* __restrict__ x1,
                                                   const float* __restrict__ x3,
                                                   __bf16* __restrict__ gated, int n) {
  const int i = blockIdx.x * 256 + threadIdx.x;
  if (i < n) {
    const float a = x1[i];
    const float sig = 1.0f / (1.0f + __expf(-a));
    gated[i] = (__bf16)(sig * a * x3[i]);
  }
}

__global__ __launch_bounds__(256) void cvt_f32_bf16(const float* __restrict__ src,
                                                    __bf16* __restrict__ dst, int n) {
  for (int i = blockIdx.x * 256 + threadIdx.x; i < n; i += gridDim.x * 256)
    dst[i] = (__bf16)src[i];
}

// ---------------------------------------------------------------------------
extern "C" void kernel_launch(void* const* d_in, const int* in_sizes, int n_in,
                              void* d_out, int out_size, void* d_ws, size_t ws_size,
                              hipStream_t stream) {
  (void)in_sizes; (void)n_in; (void)out_size; (void)ws_size;
  const float* x    = (const float*)d_in[0];
  const float* Wqkv = (const float*)d_in[1];
  const float* Wo   = (const float*)d_in[2];
  const float* g1   = (const float*)d_in[3];
  const float* g2   = (const float*)d_in[4];
  const float* w1   = (const float*)d_in[5];
  const float* w2   = (const float*)d_in[6];
  const float* w3   = (const float*)d_in[7];

  size_t off = 0;
  auto alloc = [&](size_t bytes) -> void* {
    void* p = (char*)d_ws + off;
    off += (bytes + 255) & ~(size_t)255;
    return p;
  };

  __bf16* Wqkv_b = (__bf16*)alloc((size_t)3 * DMODEL * DMODEL * 2);
  __bf16* Wo_b   = (__bf16*)alloc((size_t)DMODEL * DMODEL * 2);
  __bf16* w1_b   = (__bf16*)alloc((size_t)DFF * DMODEL * 2);
  __bf16* w2_b   = (__bf16*)alloc((size_t)DMODEL * DFF * 2);
  __bf16* w3_b   = (__bf16*)alloc((size_t)DFF * DMODEL * 2);
  __bf16* h1     = (__bf16*)alloc((size_t)MROWS * DMODEL * 2);
  float*  qkv    = (float*) alloc((size_t)MROWS * 3 * DMODEL * 4);
  __bf16* qTb    = (__bf16*)alloc((size_t)MROWS * DMODEL * 2);
  __bf16* kTb    = (__bf16*)alloc((size_t)MROWS * DMODEL * 2);
  __bf16* vTb    = (__bf16*)alloc((size_t)MROWS * DMODEL * 2);
  __bf16* attnb  = (__bf16*)alloc((size_t)MROWS * DMODEL * 2);
  float*  x2     = (float*) alloc((size_t)MROWS * DMODEL * 4);
  __bf16* h2     = (__bf16*)alloc((size_t)MROWS * DMODEL * 2);
  float*  x1b    = (float*) alloc((size_t)MROWS * DFF * 4);
  float*  x3b    = (float*) alloc((size_t)MROWS * DFF * 4);
  __bf16* gatedb = (__bf16*)alloc((size_t)MROWS * DFF * 2);

  // Weights -> bf16 (deterministic, every call)
  cvt_f32_bf16<<<2048, 256, 0, stream>>>(Wqkv, Wqkv_b, 3 * DMODEL * DMODEL);
  cvt_f32_bf16<<<2048, 256, 0, stream>>>(Wo,   Wo_b,   DMODEL * DMODEL);
  cvt_f32_bf16<<<2048, 256, 0, stream>>>(w1,   w1_b,   DFF * DMODEL);
  cvt_f32_bf16<<<2048, 256, 0, stream>>>(w2,   w2_b,   DMODEL * DFF);
  cvt_f32_bf16<<<2048, 256, 0, stream>>>(w3,   w3_b,   DFF * DMODEL);

  // h1 = rmsnorm(x, g1)
  rmsnorm_bf16<<<MROWS, 256, 0, stream>>>(x, g1, h1);

  // qkv = h1 @ Wqkv^T
  gemm_bf16<0><<<dim3(3 * DMODEL / BN, MROWS / BM), 256, GEMM_LDS_BYTES, stream>>>(
      h1, Wqkv_b, qkv, nullptr, MROWS, 3 * DMODEL, DMODEL);

  // RoPE(q,k) + scatter to [B,H,S,64] bf16
  rope_scatter<<<(BATCH * SEQ * NHEADS * 32) / 256, 256, 0, stream>>>(qkv, qTb, kTb, vTb);

  // causal flash attention
  flash_attn<<<BATCH * NHEADS * (SEQ / 128), 256, ATTN_LDS_BYTES, stream>>>(
      qTb, kTb, vTb, attnb);

  // x2 = x + attn @ Wo^T
  gemm_bf16<1><<<dim3(DMODEL / BN, MROWS / BM), 256, GEMM_LDS_BYTES, stream>>>(
      attnb, Wo_b, x2, x, MROWS, DMODEL, DMODEL);

  // h2 = rmsnorm(x2, g2)
  rmsnorm_bf16<<<MROWS, 256, 0, stream>>>(x2, g2, h2);

  // x1 = h2 @ w1^T ; x3 = h2 @ w3^T
  gemm_bf16<0><<<dim3(DFF / BN, MROWS / BM), 256, GEMM_LDS_BYTES, stream>>>(
      h2, w1_b, x1b, nullptr, MROWS, DFF, DMODEL);
  gemm_bf16<0><<<dim3(DFF / BN, MROWS / BM), 256, GEMM_LDS_BYTES, stream>>>(
      h2, w3_b, x3b, nullptr, MROWS, DFF, DMODEL);

  // gated = silu(x1) * x3
  swiglu_gate<<<(MROWS * DFF) / 256, 256, 0, stream>>>(x1b, x3b, gatedb, MROWS * DFF);

  // out = x2 + gated @ w2^T
  gemm_bf16<1><<<dim3(DMODEL / BN, MROWS / BM), 256, GEMM_LDS_BYTES, stream>>>(
      gatedb, w2_b, (float*)d_out, x2, MROWS, DMODEL, DFF);
}
